// RecurrentCellWrapper_45595372814658
// MI455X (gfx1250) — matrix operand order repaired and checked by reading, not compiled, not run on hardware
//
#include <hip/hip_runtime.h>
#include <hip/hip_bf16.h>
#include <stdint.h>

// GRU: B=64, S=512, I=H=1024, gates (r,z,n), returns final hidden [B,H] fp32.
#define B_   64
#define S_   512
#define I_   1024
#define H_   1024
#define G3H  3072

typedef __attribute__((ext_vector_type(16))) __bf16        bf16x16;
typedef __attribute__((ext_vector_type(8)))  float          f32x8;
typedef __attribute__((ext_vector_type(8)))  unsigned short u16x8;
typedef __attribute__((ext_vector_type(16))) unsigned short u16x16;
typedef __attribute__((ext_vector_type(4)))  unsigned int   u32x4;
typedef __attribute__((ext_vector_type(4)))  int            i32x4;
typedef __attribute__((ext_vector_type(8)))  int            i32x8;

struct u16x8x2 { u16x8 lo, hi; };

__device__ __forceinline__ unsigned short f32_to_bf16(float f) {
    union { float f; unsigned int u; } v; v.f = f;
    unsigned int lsb = (v.u >> 16) & 1u;
    return (unsigned short)((v.u + 0x7FFFu + lsb) >> 16);  // RNE
}

__device__ __forceinline__ float sigmoidf_(float x) {
    return 1.0f / (1.0f + __expf(-x));
}

// ---------------------------------------------------------- TDM 2D tile load
// Loads a [tile_rows x tile_k] bf16 tile (row stride = row_stride elements)
// from global memory into contiguous LDS at lds_byte_addr. ISA §8.3/8.4 D#.
#define HAVE_TDM __has_builtin(__builtin_amdgcn_tensor_load_to_lds)

__device__ __forceinline__ void tdm_load_tile_2d(unsigned int lds_byte_addr,
                                                 unsigned long long gaddr,
                                                 int tile_k, int tile_rows,
                                                 int row_stride, int tensor_k) {
#if HAVE_TDM
    u32x4 g0;
    g0[0] = 1u;                                           // count=1, load, user D#
    g0[1] = lds_byte_addr;                                // [63:32]  lds_addr
    g0[2] = (unsigned int)(gaddr & 0xFFFFFFFFull);        // [95:64]  global_addr lo
    g0[3] = (unsigned int)((gaddr >> 32) & 0x01FFFFFFull) // [120:96] global_addr hi
            | (2u << 30);                                 // [127:126] type=2 (image)
    i32x8 g1;
    g1[0] = (1 << 16);                                    // data_size=1 -> 2 bytes
    g1[1] = (tensor_k & 0xFFFF) << 16;                    // tensor_dim0[15:0]
    g1[2] = ((tensor_k >> 16) & 0xFFFF)                   // tensor_dim0[31:16]
            | ((tile_rows & 0xFFFF) << 16);               // tensor_dim1[15:0]
    g1[3] = (tile_k & 0xFFFF) << 16;                      // tile_dim0
    g1[4] = (tile_rows & 0xFFFF);                         // tile_dim1
    g1[5] = row_stride;                                   // tensor_dim0_stride lo
    g1[6] = 0;
    g1[7] = 0;
    i32x4 gz = {0, 0, 0, 0};                              // groups 2/3 unused (2D)
#if defined(__clang_major__) && (__clang_major__ >= 23)
    i32x8 gz8 = {0, 0, 0, 0, 0, 0, 0, 0};
    __builtin_amdgcn_tensor_load_to_lds(g0, g1, gz, gz, gz8, 0);
#else
    __builtin_amdgcn_tensor_load_to_lds(g0, g1, gz, gz, 0);
#endif
#endif
}

// ---------------------------------------------------------------- converts
__global__ void cvt_f32_to_bf16(const float* __restrict__ src,
                                unsigned short* __restrict__ dst, int n) {
    int i = blockIdx.x * blockDim.x + threadIdx.x;
    int stride = gridDim.x * blockDim.x;
    for (; i < n; i += stride) dst[i] = f32_to_bf16(src[i]);
}

__global__ void init_h(const float* __restrict__ h0,
                       float* __restrict__ hf,
                       unsigned short* __restrict__ hb, int n) {
    int i = blockIdx.x * blockDim.x + threadIdx.x;
    int stride = gridDim.x * blockDim.x;
    for (; i < n; i += stride) {
        float v = h0[i];
        hf[i] = v;
        hb[i] = f32_to_bf16(v);
    }
}

// ------------------------------------------------- phase 2: gi = X @ Wih^T + b_ih
// M = B*S = 32768, N = 3H = 3072, K = I = 1024.
// block = 256 threads (8 waves), block tile 32(M) x 128(N).
// A (32 x 128K chunk) is DMA'd by the TDM into double-buffered LDS; each wave
// keeps TWO 16x16 accumulators (M rows 0-15 and 16-31) sharing one B register
// tile, so each 32B/lane global B load feeds two WMMAs.
#define KCH 128                       // K chunk per TDM tile
#define NKC (I_ / KCH)                // 8 chunks
#define MT_ 32                        // M rows per block

__global__ __launch_bounds__(256)
void gi_gemm(const unsigned short* __restrict__ Xb,    // [M, K] bf16 row-major
             const unsigned short* __restrict__ Wb,    // [N, K] bf16 row-major
             const float*          __restrict__ bias,  // [N]
             float*                __restrict__ gi) {  // [M, N] fp32
    __shared__ unsigned short Abuf[2][MT_ * KCH];      // 2 x 8 KB staging

    const int lane = threadIdx.x & 31;
    const int wave = threadIdx.x >> 5;                 // 0..7
    const int m0   = blockIdx.y * MT_;
    const int n0   = blockIdx.x * 128 + wave * 16;
    const int hi   = lane >> 4;                        // half-wave
    const int mr   = lane & 15;

    const unsigned short* brow = Wb + (size_t)(n0 + mr) * I_;
#if HAVE_TDM
    const unsigned long long abase =
        (unsigned long long)(uintptr_t)(Xb + (size_t)m0 * I_);
    const unsigned int lds0 = (unsigned int)(uintptr_t)&Abuf[0][0];
    const unsigned int lds1 = (unsigned int)(uintptr_t)&Abuf[1][0];

    if (wave == 0)                                     // prologue: chunk 0
        tdm_load_tile_2d(lds0, abase, KCH, MT_, I_, I_);
#else
    const unsigned short* arow_g0 = Xb + (size_t)(m0 + mr) * I_;
    const unsigned short* arow_g1 = Xb + (size_t)(m0 + 16 + mr) * I_;
#endif

    f32x8 acc0 = {};                                   // M rows m0 .. m0+15
    f32x8 acc1 = {};                                   // M rows m0+16 .. m0+31
    for (int kb = 0; kb < NKC; ++kb) {
        const int cur = kb & 1;
#if HAVE_TDM
        if (wave == 0) {
            if (kb + 1 < NKC) {
                tdm_load_tile_2d(cur ? lds0 : lds1,
                                 abase + (unsigned long long)(kb + 1) * KCH * 2,
                                 KCH, MT_, I_, I_);
                __builtin_amdgcn_s_wait_tensorcnt(1);  // chunk kb landed
            } else {
                __builtin_amdgcn_s_wait_tensorcnt(0);
            }
        }
        __syncthreads();                               // Abuf[cur] visible to all
        const unsigned short* arow0 = &Abuf[cur][mr * KCH];
        const unsigned short* arow1 = &Abuf[cur][(16 + mr) * KCH];
#endif
#pragma unroll
        for (int ki = 0; ki < KCH; ki += 32) {
            const int kg = kb * KCH + ki;              // global K for B
#if !HAVE_TDM
            const unsigned short* arow0 = arow_g0 + kb * KCH;
            const unsigned short* arow1 = arow_g1 + kb * KCH;
#endif
            // B tile 32x16: lane = column, 16 contiguous K at kg + hi*16
            u16x16 bp = *(const u16x16*)(brow + kg + hi * 16);
            bf16x16 b = __builtin_bit_cast(bf16x16, bp);
            __builtin_prefetch(brow + kg + 512, 0, 3);
            // A tiles 16x32: elems 0..15 = K[ki+hi*16..] contiguous
            u16x8x2 ap;
            ap.lo = *(const u16x8*)(arow0 + ki + hi * 16);
            ap.hi = *(const u16x8*)(arow0 + ki + hi * 16 + 8);
            bf16x16 a0 = __builtin_bit_cast(bf16x16, ap);
            acc0 = __builtin_amdgcn_wmma_f32_16x16x32_bf16(
                       false, a0, false, b, (short)0, acc0, false, false);
            ap.lo = *(const u16x8*)(arow1 + ki + hi * 16);
            ap.hi = *(const u16x8*)(arow1 + ki + hi * 16 + 8);
            bf16x16 a1 = __builtin_bit_cast(bf16x16, ap);
            acc1 = __builtin_amdgcn_wmma_f32_16x16x32_bf16(
                       false, a1, false, b, (short)0, acc1, false, false);
        }
#if HAVE_TDM
        __syncthreads();                               // done reading Abuf[cur]
#endif
    }

    const int n  = n0 + mr;
    const float bv = bias[n];
#pragma unroll
    for (int r = 0; r < 8; ++r) {
        const int m = m0 + r + hi * 8;                 // C layout: M = r + 8*(lane>=16)
        gi[(size_t)m * G3H + n]        = acc0[r] + bv;
        gi[(size_t)(m + 16) * G3H + n] = acc1[r] + bv;
    }
}

// ------------------------------------------------- phase 3: one GRU time step
// grid = (H/16, B/16); block = 96 (3 waves, one per gate r/z/n).
__global__ __launch_bounds__(96)
void gru_step(const unsigned short* __restrict__ hb_in,  // [B,H] bf16
              const float*          __restrict__ hf_in,  // [B,H] f32
              const unsigned short* __restrict__ Whh,    // [3H,H] bf16 row-major
              const float*          __restrict__ b_hh,   // [3H]
              const float*          __restrict__ gi,     // [B*S, 3H] f32
              int s,
              float*          __restrict__ hf_out,       // [B,H] f32 (or d_out)
              unsigned short* __restrict__ hb_out) {     // [B,H] bf16
    __shared__ float lds[3][256];

    const int lane = threadIdx.x & 31;
    const int wave = threadIdx.x >> 5;                   // gate 0=r,1=z,2=n
    const int nt   = blockIdx.x;                         // hidden tile
    const int mt   = blockIdx.y;                         // batch tile
    const int hi   = lane >> 4;
    const int mr   = lane & 15;

    const int gcol0 = wave * H_ + nt * 16;
    const unsigned short* arow = hb_in + (size_t)(mt * 16 + mr) * H_;
    const unsigned short* wrow = Whh + (size_t)(gcol0 + mr) * H_;

    f32x8 acc = {};
    for (int k = 0; k < H_; k += 32) {
        u16x8x2 ap;
        ap.lo = *(const u16x8*)(arow + k + hi * 16);
        ap.hi = *(const u16x8*)(arow + k + hi * 16 + 8);
        bf16x16 a = __builtin_bit_cast(bf16x16, ap);
        u16x16 bp = *(const u16x16*)(wrow + k + hi * 16);
        bf16x16 b = __builtin_bit_cast(bf16x16, bp);
        __builtin_prefetch(wrow + k + 512, 0, 3);
        acc = __builtin_amdgcn_wmma_f32_16x16x32_bf16(
                  false, a, false, b, (short)0, acc, false, false);
    }

    const float bv = b_hh[gcol0 + mr];                   // bias of this column
#pragma unroll
    for (int r = 0; r < 8; ++r) {
        const int row = r + hi * 8;                      // batch row in tile
        lds[wave][row * 16 + mr] = acc[r] + bv;          // gh_g + b_hh_g
    }
    __syncthreads();

    // fuse gates: 256 elements over 96 threads
    for (int idx = threadIdx.x; idx < 256; idx += 96) {
        const int brw  = idx >> 4;
        const int j    = idx & 15;
        const int b    = mt * 16 + brw;
        const int hcol = nt * 16 + j;
        const size_t gbase = ((size_t)b * S_ + s) * G3H;

        const float ir = gi[gbase + hcol];
        const float iz = gi[gbase + H_ + hcol];
        const float in = gi[gbase + 2 * H_ + hcol];

        const float r_ = sigmoidf_(ir + lds[0][idx]);
        const float z_ = sigmoidf_(iz + lds[1][idx]);
        const float n_ = tanhf(in + r_ * lds[2][idx]);

        const float ho = hf_in[(size_t)b * H_ + hcol];
        const float hn = (1.0f - z_) * n_ + z_ * ho;

        hf_out[(size_t)b * H_ + hcol] = hn;
        hb_out[(size_t)b * H_ + hcol] = f32_to_bf16(hn);
    }
}

// ---------------------------------------------------------------- launcher
extern "C" void kernel_launch(void* const* d_in, const int* in_sizes, int n_in,
                              void* d_out, int out_size, void* d_ws, size_t ws_size,
                              hipStream_t stream) {
    const float* x    = (const float*)d_in[0];   // [B,S,I]
    const float* h0   = (const float*)d_in[1];   // [B,H]
    const float* W_ih = (const float*)d_in[2];   // [3H,I]
    const float* W_hh = (const float*)d_in[3];   // [3H,H]
    const float* b_ih = (const float*)d_in[4];   // [3H]
    const float* b_hh = (const float*)d_in[5];   // [3H]
    float* out = (float*)d_out;                  // [B,H]

    char* ws = (char*)d_ws;
    size_t off = 0;
    auto alloc = [&](size_t bytes) -> void* {
        void* p = ws + off;
        off = (off + bytes + 255) & ~(size_t)255;
        return p;
    };
    unsigned short* xb   = (unsigned short*)alloc((size_t)B_ * S_ * I_ * 2);
    unsigned short* wihb = (unsigned short*)alloc((size_t)G3H * I_ * 2);
    unsigned short* whhb = (unsigned short*)alloc((size_t)G3H * H_ * 2);
    float*          gi   = (float*)alloc((size_t)B_ * S_ * G3H * 4);
    float*          hf0  = (float*)alloc((size_t)B_ * H_ * 4);
    float*          hf1  = (float*)alloc((size_t)B_ * H_ * 4);
    unsigned short* hb0  = (unsigned short*)alloc((size_t)B_ * H_ * 2);
    unsigned short* hb1  = (unsigned short*)alloc((size_t)B_ * H_ * 2);
    float*          hf[2] = { hf0, hf1 };
    unsigned short* hb[2] = { hb0, hb1 };

    // phase 1: bf16 conversions + h init
    {
        int n = B_ * S_ * I_;
        cvt_f32_to_bf16<<<dim3(2048), dim3(256), 0, stream>>>(x, xb, n);
        n = G3H * I_;
        cvt_f32_to_bf16<<<dim3(512), dim3(256), 0, stream>>>(W_ih, wihb, n);
        n = G3H * H_;
        cvt_f32_to_bf16<<<dim3(512), dim3(256), 0, stream>>>(W_hh, whhb, n);
        n = B_ * H_;
        init_h<<<dim3(64), dim3(256), 0, stream>>>(h0, hf[0], hb[0], n);
    }

    // phase 2: gi = X @ Wih^T + b_ih  (M=32768, N=3072, K=1024)
    {
        dim3 grid(G3H / 128, (B_ * S_) / MT_);
        gi_gemm<<<grid, dim3(256), 0, stream>>>(xb, wihb, b_ih, gi);
    }

    // phase 3: sequential recurrence (512 step kernels, graph-captured)
    for (int s = 0; s < S_; ++s) {
        const int cur = s & 1;
        const int nxt = cur ^ 1;
        float* hfo = (s == S_ - 1) ? out : hf[nxt];
        gru_step<<<dim3(H_ / 16, B_ / 16), dim3(96), 0, stream>>>(
            hb[cur], hf[cur], whhb, b_hh, gi, s, hfo, hb[nxt]);
    }
}
